// PrototypeMemoryModule_91139206021646
// MI455X (gfx1250) — matrix operand order repaired and checked by reading, not compiled, not run on hardware
//
#include <hip/hip_runtime.h>

// ---------------------------------------------------------------------------
// PrototypeMemoryModule for MI455X (gfx1250, wave32, WMMA)
//   B=2048 queries, P=65536 prototypes, D=1024, L=32 labels, top-8, T=0.1
// Pipeline:
//   1) l2norm_bf16_kernel : fp32 rows -> L2-normalized bf16 rows (Q and bank)
//   2) gemm_topk_kernel   : bf16 WMMA GEMM (16x64 per wave, SW-pipelined K)
//                           + fused per-(row,strip) top-8 partials -> ws
//   3) merge_topk_kernel  : merge 256 partials/row, softmax(/0.1),
//                           gather labels, write prior[2048,32] + stats[2048,2]
// ---------------------------------------------------------------------------

typedef __bf16 bf16_t;
typedef bf16_t v16bf __attribute__((ext_vector_type(16)));
typedef bf16_t v8bf  __attribute__((ext_vector_type(8)));
typedef float  v8f   __attribute__((ext_vector_type(8)));

union V16 { v16bf v; v8bf h[2]; };

#define B_ROWS 2048
#define P_ROWS 65536
#define DIM    1024
#define L_DIM  32
#define NCHUNK 64      // split P into 64 chunks of 1024 columns
#define NSUB   4       // 4 column-strips of 64 within the 256-wide block tile
#define TOPK   8
#define BN     256     // block tile columns per iteration
#define CPAD   260     // LDS row stride in floats

__device__ __forceinline__ unsigned short f2bf(float f) {
  // round-to-nearest-even fp32 -> bf16
  unsigned int u = __float_as_uint(f);
  u += 0x7FFFu + ((u >> 16) & 1u);
  return (unsigned short)(u >> 16);
}

// ---------------------------------------------------------------------------
// Kernel 1: per-row L2 normalize (x / max(||x||, eps)), emit bf16.
// ---------------------------------------------------------------------------
__global__ __launch_bounds__(256) void l2norm_bf16_kernel(
    const float* __restrict__ in, unsigned short* __restrict__ out) {
  const int row = blockIdx.x;
  const int t = threadIdx.x;
  const float4 v = ((const float4*)(in + (size_t)row * DIM))[t];
  __shared__ float red[256];
  red[t] = v.x * v.x + v.y * v.y + v.z * v.z + v.w * v.w;
  __syncthreads();
  #pragma unroll
  for (int s = 128; s > 0; s >>= 1) {
    if (t < s) red[t] += red[t + s];
    __syncthreads();
  }
  const float inv = 1.0f / fmaxf(sqrtf(red[0]), 1e-12f);
  uint2 o;
  o.x = (unsigned)f2bf(v.x * inv) | ((unsigned)f2bf(v.y * inv) << 16);
  o.y = (unsigned)f2bf(v.z * inv) | ((unsigned)f2bf(v.w * inv) << 16);
  ((uint2*)(out + (size_t)row * DIM))[t] = o;
}

// ---------------------------------------------------------------------------
// Kernel 2: fused WMMA GEMM + top-8 over a 1024-column chunk of P.
// grid = (B/32, 64). Block = 256 threads = 8 waves arranged 2(M) x 4(N);
// each wave computes a 16(M) x 64(N) sub-tile = 4 WMMA tiles, K pipelined
// 2-deep in steps of 32 with v_wmma_f32_16x16x32_bf16.
// ---------------------------------------------------------------------------
__global__ __launch_bounds__(256) void gemm_topk_kernel(
    const unsigned short* __restrict__ qbf,
    const unsigned short* __restrict__ bankbf,
    float* __restrict__ pvals, int* __restrict__ pidx) {
  __shared__ float smem[32 * CPAD];   // 32 rows x 256 cols (+4 pad)

  const int tid  = threadIdx.x;
  const int lane = tid & 31;
  const int wave = tid >> 5;
  const int mi = wave & 1;           // which 16-row half of the 32-row tile
  const int nw = wave >> 1;          // which 64-col group of the 256-col tile

  const int row0  = blockIdx.x * 32;
  const int pbase = blockIdx.y * 1024;

  // A fragment addressing (ISA 16-bit A 16x32 layout)
  const int arow  = row0 + mi * 16 + (lane & 15);
  const int akoff = (lane >> 4) * 8;
  const unsigned short* aptr = qbf + (size_t)arow * DIM + akoff;

  // B fragment addressing (bank rows are B columns, contiguous K)
  const int lcol  = lane & 15;
  const int bkoff = (lane >> 4) * 8;

  // top-8 state: waves 0-3 (tid<128) each own (row = tid>>2, strip = tid&3)
  float tv[TOPK];
  int   ti[TOPK];
  #pragma unroll
  for (int j = 0; j < TOPK; ++j) { tv[j] = -3.0e38f; ti[j] = 0; }

  for (int iter = 0; iter < 1024 / BN; ++iter) {   // 4 x 256 columns = 1024
    const int colbase = pbase + iter * BN;
    v8f acc[4];
    #pragma unroll
    for (int f = 0; f < 4; ++f) acc[f] = (v8f){};

    const unsigned short* bptr =
        bankbf + (size_t)(colbase + nw * 64 + lcol) * DIM + bkoff;

    // 2-deep software pipeline over K
    V16 a[2], b[4][2];
    a[0].h[0] = *(const v8bf*)(aptr);
    a[0].h[1] = *(const v8bf*)(aptr + 16);
    #pragma unroll
    for (int f = 0; f < 4; ++f) {
      const unsigned short* bp = bptr + (size_t)(f * 16) * DIM;
      b[f][0].h[0] = *(const v8bf*)(bp);
      b[f][0].h[1] = *(const v8bf*)(bp + 16);
    }

    #pragma unroll 2
    for (int kt = 0; kt < 32; ++kt) {
      const int cur = kt & 1;
      const int nxt = cur ^ 1;
      const int kn = kt * 32 + 32;   // one-step overrun at kt=31 stays in ws
      a[nxt].h[0] = *(const v8bf*)(aptr + kn);
      a[nxt].h[1] = *(const v8bf*)(aptr + kn + 16);
      #pragma unroll
      for (int f = 0; f < 4; ++f) {
        const unsigned short* bp = bptr + (size_t)(f * 16) * DIM + kn;
        b[f][nxt].h[0] = *(const v8bf*)(bp);
        b[f][nxt].h[1] = *(const v8bf*)(bp + 16);
      }
      #pragma unroll
      for (int f = 0; f < 4; ++f) {
        acc[f] = __builtin_amdgcn_wmma_f32_16x16x32_bf16(
            false, a[cur].v, false, b[f][cur].v, (short)0, acc[f], false, false);
      }
    }

    __syncthreads();   // previous iteration's top-k scan is done
    // C layout: lanes 0-15 -> M=r, N=lane; lanes 16-31 -> M=r+8, N=lane-16
    const int crow  = mi * 16 + (lane >> 4) * 8;
    const int ccol0 = nw * 64 + lcol;
    #pragma unroll
    for (int f = 0; f < 4; ++f) {
      #pragma unroll
      for (int r = 0; r < 8; ++r) {
        smem[(crow + r) * CPAD + ccol0 + f * 16] = acc[f][r];
      }
    }
    __syncthreads();

    if (tid < 128) {                 // waves 0-3: wave-uniform branch
      const int srow = tid >> 2;     // 0..31
      const int sub  = tid & 3;      // 0..3 -> 64-col strip
      const float* rowc = smem + srow * CPAD + sub * 64;
      const int idx0 = colbase + sub * 64;
      for (int c = 0; c < 64; ++c) {
        const float v = rowc[c];
        if (v > tv[TOPK - 1]) {
          tv[TOPK - 1] = v;
          ti[TOPK - 1] = idx0 + c;
          #pragma unroll
          for (int j = TOPK - 1; j > 0; --j) {
            if (tv[j] > tv[j - 1]) {
              float x = tv[j]; tv[j] = tv[j - 1]; tv[j - 1] = x;
              int   y = ti[j]; ti[j] = ti[j - 1]; ti[j - 1] = y;
            }
          }
        }
      }
    }
  }

  if (tid < 128) {
    const int row = row0 + (tid >> 2);
    const int sub = tid & 3;
    const size_t off = (((size_t)row * NCHUNK + blockIdx.y) * NSUB + sub) * TOPK;
    #pragma unroll
    for (int j = 0; j < TOPK; ++j) {
      pvals[off + j] = tv[j];
      pidx[off + j]  = ti[j];
    }
  }
}

// ---------------------------------------------------------------------------
// Kernel 3: per-row merge of 256 sorted 8-candidate partials -> global top-8,
// softmax(values/0.1), weighted label gather, stats (max, mean).
// ---------------------------------------------------------------------------
__global__ __launch_bounds__(256) void merge_topk_kernel(
    const float* __restrict__ pvals, const int* __restrict__ pidx,
    const float* __restrict__ labels, float* __restrict__ out) {
  const int row = blockIdx.x * blockDim.x + threadIdx.x;
  if (row >= B_ROWS) return;

  float tv[TOPK];
  int   ti[TOPK];
  #pragma unroll
  for (int j = 0; j < TOPK; ++j) { tv[j] = -3.0e38f; ti[j] = 0; }

  const int ncand = NCHUNK * NSUB * TOPK;   // 2048 candidates
  const float* pv = pvals + (size_t)row * ncand;
  const int*   pi = pidx  + (size_t)row * ncand;
  for (int c = 0; c < ncand; ++c) {
    const float v = pv[c];
    if (v > tv[TOPK - 1]) {
      tv[TOPK - 1] = v;
      ti[TOPK - 1] = pi[c];
      #pragma unroll
      for (int j = TOPK - 1; j > 0; --j) {
        if (tv[j] > tv[j - 1]) {
          float x = tv[j]; tv[j] = tv[j - 1]; tv[j - 1] = x;
          int   y = ti[j]; ti[j] = ti[j - 1]; ti[j - 1] = y;
        }
      }
    }
  }

  const float vmax = tv[0];
  float vsum = 0.0f;
  float e[TOPK], es = 0.0f;
  #pragma unroll
  for (int j = 0; j < TOPK; ++j) {
    vsum += tv[j];
    e[j] = __expf((tv[j] - vmax) * 10.0f);   // / TEMPERATURE(0.1)
    es += e[j];
  }
  const float invs = 1.0f / es;

  float prior[L_DIM];
  #pragma unroll
  for (int l = 0; l < L_DIM; ++l) prior[l] = 0.0f;
  #pragma unroll
  for (int j = 0; j < TOPK; ++j) {
    const float w = e[j] * invs;
    const float* lr = labels + (size_t)ti[j] * L_DIM;
    #pragma unroll
    for (int l = 0; l < L_DIM; ++l) prior[l] += w * lr[l];
  }

  float* po = out + (size_t)row * L_DIM;
  #pragma unroll
  for (int l = 0; l < L_DIM; ++l) po[l] = prior[l];

  out[(size_t)B_ROWS * L_DIM + row * 2 + 0] = vmax;
  out[(size_t)B_ROWS * L_DIM + row * 2 + 1] = vsum * (1.0f / TOPK);
}

// ---------------------------------------------------------------------------
extern "C" void kernel_launch(void* const* d_in, const int* in_sizes, int n_in,
                              void* d_out, int out_size, void* d_ws, size_t ws_size,
                              hipStream_t stream) {
  const float* emb    = (const float*)d_in[0];  // [2048, 1024]
  const float* proto  = (const float*)d_in[1];  // [65536, 1024]
  const float* labels = (const float*)d_in[2];  // [65536, 32]
  float* out = (float*)d_out;                   // prior [2048*32] ++ stats [2048*2]

  char* ws = (char*)d_ws;
  const size_t qOff    = 0;
  const size_t bankOff = qOff    + (size_t)B_ROWS * DIM * 2;                  //   4 MB
  const size_t pvOff   = bankOff + (size_t)P_ROWS * DIM * 2;                  // 128 MB
  const size_t piOff   = pvOff   + (size_t)B_ROWS * NCHUNK * NSUB * TOPK * 4; //  16 MB
  unsigned short* qbf    = (unsigned short*)(ws + qOff);
  unsigned short* bankbf = (unsigned short*)(ws + bankOff);
  float* pvals = (float*)(ws + pvOff);
  int*   pidx  = (int*)(ws + piOff);

  // 1) normalize + bf16 convert
  l2norm_bf16_kernel<<<B_ROWS, 256, 0, stream>>>(emb, qbf);
  l2norm_bf16_kernel<<<P_ROWS, 256, 0, stream>>>(proto, bankbf);

  // 2) WMMA GEMM + fused chunk top-8
  dim3 ggrid(B_ROWS / 32, NCHUNK);
  gemm_topk_kernel<<<ggrid, 256, 0, stream>>>(qbf, bankbf, pvals, pidx);

  // 3) merge partials, softmax, label gather, stats
  merge_topk_kernel<<<(B_ROWS + 255) / 256, 256, 0, stream>>>(pvals, pidx, labels, out);
}
